// DiffusionTransformer_77653008711712
// MI455X (gfx1250) — matrix operand order repaired and checked by reading, not compiled, not run on hardware
//
#include <hip/hip_runtime.h>
#include <hip/hip_bf16.h>

typedef __bf16 bf16_t;
typedef __attribute__((ext_vector_type(16))) __bf16 v16bf;
typedef __attribute__((ext_vector_type(8)))  __bf16 v8bf;
typedef __attribute__((ext_vector_type(8)))  float  v8f;
typedef __attribute__((ext_vector_type(4)))  float  v4f;
typedef __attribute__((ext_vector_type(4)))  int    v4i_t;

#define NTOK 1024
#define CA   768
#define CS   384
#define CZ   128
#define NHD  16
#define DHD  48
#define C2   1536

// ---- CDNA5 async global->LDS path (guarded; falls back to sync copy) ------
#if defined(__has_builtin)
#  if __has_builtin(__builtin_amdgcn_global_load_async_to_lds_b128) && \
      __has_builtin(__builtin_amdgcn_s_wait_asynccnt)
#    define USE_ASYNC_LDS 1
#  endif
#endif

#if defined(USE_ASYNC_LDS)
typedef __attribute__((address_space(1))) v4i_t* as1_v4i;
typedef __attribute__((address_space(3))) v4i_t* as3_v4i;
#endif

__device__ __forceinline__ void async_wait_all() {
#if defined(USE_ASYNC_LDS)
  __builtin_amdgcn_s_wait_asynccnt(0);
#endif
}

__device__ __forceinline__ v16bf cat8(v8bf lo, v8bf hi) {
  return __builtin_shufflevector(lo, hi, 0,1,2,3,4,5,6,7,8,9,10,11,12,13,14,15);
}
__device__ __forceinline__ float sigf(float x) { return 1.f / (1.f + __expf(-x)); }

// ---------------------------------------------------------------------------
// Transpose + convert: src f32 [R,C] -> dst bf16 [C,R]
// ---------------------------------------------------------------------------
__global__ __launch_bounds__(256) void packT_kernel(const float* __restrict__ src,
                                                    bf16_t* __restrict__ dst,
                                                    int R, int C) {
  __shared__ float tile[32][33];
  const int tx = threadIdx.x & 31, ty = threadIdx.x >> 5; // 32 x 8
  const int c0 = blockIdx.x * 32, r0 = blockIdx.y * 32;
  #pragma unroll
  for (int i = 0; i < 4; ++i) {
    int r = r0 + ty + i * 8;
    if (r < R && (c0 + tx) < C) tile[ty + i * 8][tx] = src[(size_t)r * C + c0 + tx];
  }
  __syncthreads();
  #pragma unroll
  for (int i = 0; i < 4; ++i) {
    int cc = c0 + ty + i * 8, rr = r0 + tx;
    if (cc < C && rr < R) dst[(size_t)cc * R + rr] = (bf16_t)tile[tx][ty + i * 8];
  }
}

// ---------------------------------------------------------------------------
// Row LayerNorm * g + b : x [rows,C] -> out [rows,C]
// ---------------------------------------------------------------------------
__global__ __launch_bounds__(128) void ln_scale_kernel(const float* __restrict__ x,
                                                       const float* __restrict__ g,
                                                       const float* __restrict__ b,
                                                       float* __restrict__ out, int C) {
  const int row = blockIdx.x;
  const float* xr = x + (size_t)row * C;
  float s1 = 0.f, s2 = 0.f;
  for (int c = threadIdx.x; c < C; c += 128) { float v = xr[c]; s1 += v; s2 += v * v; }
  #pragma unroll
  for (int m = 1; m < 32; m <<= 1) { s1 += __shfl_xor(s1, m, 32); s2 += __shfl_xor(s2, m, 32); }
  __shared__ float r1[4], r2[4];
  const int wave = threadIdx.x >> 5, lane = threadIdx.x & 31;
  if (lane == 0) { r1[wave] = s1; r2[wave] = s2; }
  __syncthreads();
  s1 = r1[0] + r1[1] + r1[2] + r1[3];
  s2 = r2[0] + r2[1] + r2[2] + r2[3];
  const float mean = s1 / C;
  const float rstd = rsqrtf(s2 / C - mean * mean + 1e-5f);
  float* o = out + (size_t)row * C;
  for (int c = threadIdx.x; c < C; c += 128) o[c] = (xr[c] - mean) * rstd * g[c] + b[c];
}

// ---------------------------------------------------------------------------
// adaLN combine: out = sigmoid(G) * LN(a) + S   (row-wise LN over C)
// ---------------------------------------------------------------------------
__global__ __launch_bounds__(256) void adaln_combine_kernel(const float* __restrict__ a,
                                                            const float* __restrict__ G,
                                                            const float* __restrict__ S,
                                                            float* __restrict__ out, int C) {
  const int row = blockIdx.x;
  const float* xr = a + (size_t)row * C;
  float s1 = 0.f, s2 = 0.f;
  for (int c = threadIdx.x; c < C; c += 256) { float v = xr[c]; s1 += v; s2 += v * v; }
  #pragma unroll
  for (int m = 1; m < 32; m <<= 1) { s1 += __shfl_xor(s1, m, 32); s2 += __shfl_xor(s2, m, 32); }
  __shared__ float r1[8], r2[8];
  const int wave = threadIdx.x >> 5, lane = threadIdx.x & 31;
  if (lane == 0) { r1[wave] = s1; r2[wave] = s2; }
  __syncthreads();
  s1 = 0.f; s2 = 0.f;
  #pragma unroll
  for (int w = 0; w < 8; ++w) { s1 += r1[w]; s2 += r2[w]; }
  const float mean = s1 / C;
  const float rstd = rsqrtf(s2 / C - mean * mean + 1e-5f);
  const size_t base = (size_t)row * C;
  for (int c = threadIdx.x; c < C; c += 256) {
    float ln = (xr[c] - mean) * rstd;
    out[base + c] = sigf(G[base + c]) * ln + S[base + c];
  }
}

// ---------------------------------------------------------------------------
// Generic WMMA GEMM: C[M,Nt] = epilogue(A[M,K] @ Wt[Nt,K]^T + bias)
// A: f32 row-major, Wt: bf16 pre-transposed [Nt,K].
// Block tile 64x128, 8 waves (2x4), wave tile 32x32 (4x v_wmma bf16/k-step).
// Double-buffered LDS; Wt tiles staged with global_load_async_to_lds_b128
// (ASYNCcnt) when available; A tile loads issued one iteration early.
// emode: 0 f32, 1 sigmoid, 2 bf16, 3 aux2+aux1*acc, 4 silu, 5 aux1*acc
// ---------------------------------------------------------------------------
#define TBM 64
#define TBN 128
#define TBK 32

__device__ __forceinline__ void stage_w(const bf16_t* __restrict__ Wt, bf16_t* dstbuf,
                                        int bn0, int K, int k0, int tid) {
  #pragma unroll
  for (int it = 0; it < 2; ++it) {
    int idx = tid + it * 256;
    int r = idx >> 2, c8 = (idx & 3) << 3;
    const bf16_t* src = Wt + (size_t)(bn0 + r) * K + k0 + c8;
    bf16_t* dst = dstbuf + r * TBK + c8;
#if defined(USE_ASYNC_LDS)
    __builtin_amdgcn_global_load_async_to_lds_b128(
        (as1_v4i)(uintptr_t)src,
        (as3_v4i)(unsigned)(uintptr_t)dst,
        0, 0);
#else
    *(v8bf*)dst = *(const v8bf*)src;
#endif
  }
}

__global__ __launch_bounds__(256) void gemm_kernel(
    const float* __restrict__ A, const bf16_t* __restrict__ Wt,
    const float* __restrict__ bias, float* __restrict__ outF, bf16_t* __restrict__ outB,
    const float* __restrict__ aux1, const float* __restrict__ aux2,
    int M, int K, int Nt, int emode) {
  __shared__ __align__(32) bf16_t As[2][TBM * TBK];
  __shared__ __align__(32) bf16_t Wls[2][TBN * TBK];
  const int tid = threadIdx.x;
  const int lane = tid & 31, wave = tid >> 5;
  const int wm = wave >> 2, wn = wave & 3;
  const int bm0 = blockIdx.y * TBM, bn0 = blockIdx.x * TBN;
  const int fm = lane & 15, hi = lane >> 4;
  const int kbA = hi * 8, kbB = hi * 16;
  const int ra0 = tid >> 3,         ca0 = (tid & 7) << 2;      // A-stage slot 0
  const int ra1 = (tid + 256) >> 3, ca1 = ((tid + 256) & 7) << 2;
  const int nkt = K / TBK;
  v8f acc[2][2] = {{v8f{}, v8f{}}, {v8f{}, v8f{}}};

  // prologue: stage tile 0 into buffer 0
  stage_w(Wt, &Wls[0][0], bn0, K, 0, tid);
  {
    v4f v0 = *(const v4f*)(A + (size_t)(bm0 + ra0) * K + ca0);
    v4f v1 = *(const v4f*)(A + (size_t)(bm0 + ra1) * K + ca1);
    bf16_t* d0 = &As[0][ra0 * TBK + ca0];
    d0[0] = (bf16_t)v0.x; d0[1] = (bf16_t)v0.y; d0[2] = (bf16_t)v0.z; d0[3] = (bf16_t)v0.w;
    bf16_t* d1 = &As[0][ra1 * TBK + ca1];
    d1[0] = (bf16_t)v1.x; d1[1] = (bf16_t)v1.y; d1[2] = (bf16_t)v1.z; d1[3] = (bf16_t)v1.w;
  }
  async_wait_all();
  __syncthreads();

  for (int t = 0; t < nkt; ++t) {
    const int b = t & 1;
    const bool more = (t + 1) < nkt;
    v4f apre0, apre1;
    if (more) {
      const int k1 = (t + 1) * TBK;
      stage_w(Wt, &Wls[b ^ 1][0], bn0, K, k1, tid);      // async -> LDS
      apre0 = *(const v4f*)(A + (size_t)(bm0 + ra0) * K + k1 + ca0);
      apre1 = *(const v4f*)(A + (size_t)(bm0 + ra1) * K + k1 + ca1);
    }
    // compute current buffer
    v16bf af[2], bfr[2];
    #pragma unroll
    for (int ti = 0; ti < 2; ++ti) {
      const bf16_t* p = &As[b][(wm * 32 + ti * 16 + fm) * TBK + kbA];
      af[ti] = cat8(*(const v8bf*)p, *(const v8bf*)(p + 16));
    }
    #pragma unroll
    for (int tj = 0; tj < 2; ++tj)
      bfr[tj] = cat8(*(const v8bf*)&Wls[b][(wn * 32 + tj * 16 + fm) * TBK + kbA], *(const v8bf*)&Wls[b][(wn * 32 + tj * 16 + fm) * TBK + kbA + 16]);
    #pragma unroll
    for (int ti = 0; ti < 2; ++ti)
      #pragma unroll
      for (int tj = 0; tj < 2; ++tj)
        acc[ti][tj] = __builtin_amdgcn_wmma_f32_16x16x32_bf16(
            false, af[ti], false, bfr[tj], (short)0, acc[ti][tj], false, false);
    if (more) {
      bf16_t* d0 = &As[b ^ 1][ra0 * TBK + ca0];
      d0[0] = (bf16_t)apre0.x; d0[1] = (bf16_t)apre0.y;
      d0[2] = (bf16_t)apre0.z; d0[3] = (bf16_t)apre0.w;
      bf16_t* d1 = &As[b ^ 1][ra1 * TBK + ca1];
      d1[0] = (bf16_t)apre1.x; d1[1] = (bf16_t)apre1.y;
      d1[2] = (bf16_t)apre1.z; d1[3] = (bf16_t)apre1.w;
      async_wait_all();
    }
    __syncthreads();
  }

  #pragma unroll
  for (int ti = 0; ti < 2; ++ti)
    #pragma unroll
    for (int tj = 0; tj < 2; ++tj)
      #pragma unroll
      for (int j = 0; j < 8; ++j) {
        int row = bm0 + wm * 32 + ti * 16 + j + hi * 8;
        int col = bn0 + wn * 32 + tj * 16 + fm;
        float v = acc[ti][tj][j];
        if (bias) v += bias[col];
        size_t idx = (size_t)row * Nt + col;
        if (emode == 0)      outF[idx] = v;
        else if (emode == 1) outF[idx] = sigf(v);
        else if (emode == 2) outB[idx] = (bf16_t)v;
        else if (emode == 3) outF[idx] = aux2[idx] + aux1[idx] * v;
        else if (emode == 4) outF[idx] = v * sigf(v);
        else                 outF[idx] = aux1[idx] * v;
      }
}

// ---------------------------------------------------------------------------
// zbias: per (i,j) pair: LN_128(z[i,j,:])*zg+zb then dot with wbz[:,h]
// out layout [i][j][h] f32. One wave per pair, 8 waves / block.
// ---------------------------------------------------------------------------
__global__ __launch_bounds__(256) void zbias_kernel(const float* __restrict__ z,
                                                    const float* __restrict__ zg,
                                                    const float* __restrict__ zb,
                                                    const float* __restrict__ wbz,
                                                    float* __restrict__ out) {
  __shared__ float s_wbz[CZ * NHD];
  __shared__ float s_g[CZ], s_b[CZ];
  __shared__ float s_t[8][CZ];
  const int tid = threadIdx.x;
  if (tid < CZ) { s_g[tid] = zg[tid]; s_b[tid] = zb[tid]; }
  #pragma unroll
  for (int it = 0; it < 8; ++it) s_wbz[tid + it * 256] = wbz[tid + it * 256];
  __syncthreads();
  const int wave = tid >> 5, lane = tid & 31;
  const size_t pos = (size_t)blockIdx.x * 8 + wave;
  const float* zp = z + pos * CZ;
  v4f v = *(const v4f*)(zp + lane * 4);
  float s1 = v.x + v.y + v.z + v.w;
  float s2 = v.x * v.x + v.y * v.y + v.z * v.z + v.w * v.w;
  #pragma unroll
  for (int m = 1; m < 32; m <<= 1) { s1 += __shfl_xor(s1, m, 32); s2 += __shfl_xor(s2, m, 32); }
  const float mean = s1 / CZ;
  const float rstd = rsqrtf(s2 / CZ - mean * mean + 1e-5f);
  #pragma unroll
  for (int q = 0; q < 4; ++q) {
    int c = lane * 4 + q;
    s_t[wave][c] = (v[q] - mean) * rstd * s_g[c] + s_b[c];
  }
  __syncthreads();
  if (lane < NHD) {
    float acc = 0.f;
    #pragma unroll 4
    for (int c = 0; c < CZ; ++c) acc += s_t[wave][c] * s_wbz[c * NHD + lane];
    out[pos * NHD + lane] = acc;
  }
}

// ---------------------------------------------------------------------------
// Flash attention, one wave per (16-query tile, head). DH=48 zero-padded.
// qbf/kbf: bf16 [N, CA]; vt: bf16 [CA, N] (row = h*48+dh); zbias [q][k][h] f32
// ---------------------------------------------------------------------------
__global__ __launch_bounds__(32) void attn_kernel(
    const bf16_t* __restrict__ qbf, const bf16_t* __restrict__ kbf,
    const bf16_t* __restrict__ vt, const float* __restrict__ zbias,
    const unsigned char* __restrict__ mask, const float* __restrict__ gatef,
    float* __restrict__ oheads) {
  const int qt = blockIdx.x, h = blockIdx.y;
  const int lane = threadIdx.x;
  const int fm = lane & 15, hi = lane >> 4;
  __shared__ __align__(32) bf16_t pls[256];
  const int q0 = qt * 16;
  const int kbA = hi * 8, kbB = hi * 16;
  v8bf zero8 = {};
  v16bf zero16 = {};
  const bf16_t* qrow = qbf + (size_t)(q0 + fm) * CA + h * DHD;
  v16bf qf1 = *(const v16bf*)(qrow + kbB);
  v16bf qf2 = (hi == 0) ? *(const v16bf*)(qrow + 32) : zero16;   // dh 48..63 = 0
  v8f oacc[3] = {v8f{}, v8f{}, v8f{}};
  float mrow[8], lrow[8];
  #pragma unroll
  for (int j = 0; j < 8; ++j) { mrow[j] = -1e30f; lrow[j] = 0.f; }
  const float isq = 0.14433756729740643f;   // 1/sqrt(48)

  for (int kt = 0; kt < NTOK / 16; ++kt) {
    const int k0 = kt * 16;
    if (kt + 1 < NTOK / 16) {   // hide next tile's latency behind softmax VALU
      __builtin_prefetch(kbf + (size_t)(k0 + 16 + fm) * CA + h * DHD, 0, 0);
      __builtin_prefetch(vt + (size_t)(h * DHD + fm) * NTOK + k0 + 16, 0, 0);
      __builtin_prefetch(vt + (size_t)(h * DHD + 16 + fm) * NTOK + k0 + 16, 0, 0);
      __builtin_prefetch(vt + (size_t)(h * DHD + 32 + fm) * NTOK + k0 + 16, 0, 0);
    }
    const bf16_t* krow = kbf + (size_t)(k0 + fm) * CA + h * DHD;
    v16bf kf1 = *(const v16bf*)(krow + kbB);
    v16bf kf2 = zero16;
    if (hi == 0) kf2 = *(const v16bf*)(krow + 32);            // dh 32..47
    v8f sacc = {};
    sacc = __builtin_amdgcn_wmma_f32_16x16x32_bf16(false, qf1, false, kf1, (short)0, sacc, false, false);
    sacc = __builtin_amdgcn_wmma_f32_16x16x32_bf16(false, qf2, false, kf2, (short)0, sacc, false, false);
    const float mb = mask[k0 + fm] ? 0.f : -1e9f;
    #pragma unroll
    for (int j = 0; j < 8; ++j) {
      const int m = j + hi * 8;
      float sv = sacc[j] * isq + zbias[((size_t)(q0 + m) * NTOK + (k0 + fm)) * NHD + h] + mb;
      float rm = sv;
      rm = fmaxf(rm, __shfl_xor(rm, 1, 16));
      rm = fmaxf(rm, __shfl_xor(rm, 2, 16));
      rm = fmaxf(rm, __shfl_xor(rm, 4, 16));
      rm = fmaxf(rm, __shfl_xor(rm, 8, 16));
      float mnew = fmaxf(mrow[j], rm);
      float p = __expf(sv - mnew);
      float rs = p;
      rs += __shfl_xor(rs, 1, 16);
      rs += __shfl_xor(rs, 2, 16);
      rs += __shfl_xor(rs, 4, 16);
      rs += __shfl_xor(rs, 8, 16);
      float alpha = __expf(mrow[j] - mnew);
      lrow[j] = lrow[j] * alpha + rs;
      mrow[j] = mnew;
      oacc[0][j] *= alpha; oacc[1][j] *= alpha; oacc[2][j] *= alpha;
      pls[m * 16 + fm] = (bf16_t)p;
    }
    __syncthreads();
    v16bf pf = (hi == 0) ? *(const v16bf*)&pls[fm * 16] : zero16; // K 16..31 = 0
    #pragma unroll
    for (int c = 0; c < 3; ++c) {
      v16bf vf = zero16;
      if (hi == 0) vf = *(const v16bf*)(vt + (size_t)(h * DHD + c * 16 + fm) * NTOK + k0);
      oacc[c] = __builtin_amdgcn_wmma_f32_16x16x32_bf16(false, pf, false, vf, (short)0, oacc[c], false, false);
    }
    __syncthreads();
  }
  #pragma unroll
  for (int j = 0; j < 8; ++j) {
    const float invl = 1.f / lrow[j];
    const int row = q0 + j + hi * 8;
    #pragma unroll
    for (int c = 0; c < 3; ++c) {
      const int col = h * DHD + c * 16 + fm;
      float g = gatef[(size_t)row * CA + col];
      oheads[(size_t)row * CA + col] = (oacc[c][j] * invl) * sigf(g);
    }
  }
}

// ---------------------------------------------------------------------------
extern "C" void kernel_launch(void* const* d_in, const int* in_sizes, int n_in,
                              void* d_out, int out_size, void* d_ws, size_t ws_size,
                              hipStream_t stream) {
  (void)in_sizes; (void)out_size; (void)ws_size;
  if (n_in < 30) return;
  const float* a_in = (const float*)d_in[0];
  const float* s_in = (const float*)d_in[1];
  const float* z_in = (const float*)d_in[2];
  const unsigned char* mask = (const unsigned char*)d_in[3];
  const float* p_att_sg = (const float*)d_in[4];
  const float* p_att_sb = (const float*)d_in[5];
  const float* p_att_wg = (const float*)d_in[6];
  const float* p_att_bg = (const float*)d_in[7];
  const float* p_att_ws = (const float*)d_in[8];
  const float* p_wq     = (const float*)d_in[9];
  const float* p_bq     = (const float*)d_in[10];
  const float* p_wk     = (const float*)d_in[11];
  const float* p_wv     = (const float*)d_in[12];
  const float* p_zg     = (const float*)d_in[13];
  const float* p_zb     = (const float*)d_in[14];
  const float* p_wbz    = (const float*)d_in[15];
  const float* p_wgate  = (const float*)d_in[16];
  const float* p_wo     = (const float*)d_in[17];
  const float* p_ws_out = (const float*)d_in[18];
  const float* p_bs_out = (const float*)d_in[19];
  const float* p_tr_sg  = (const float*)d_in[20];
  const float* p_tr_sb  = (const float*)d_in[21];
  const float* p_tr_wg  = (const float*)d_in[22];
  const float* p_tr_bg  = (const float*)d_in[23];
  const float* p_tr_ws  = (const float*)d_in[24];
  const float* p_w1     = (const float*)d_in[25];
  const float* p_w2     = (const float*)d_in[26];
  const float* p_w3     = (const float*)d_in[27];
  const float* p_trws_out = (const float*)d_in[28];
  const float* p_trbs_out = (const float*)d_in[29];
  float* aCur = (float*)d_out;

  char* wsb = (char*)d_ws;
  size_t off = 0;
  auto alloc = [&](size_t bytes) -> char* {
    char* p = wsb + off; off += (bytes + 255) & ~(size_t)255; return p;
  };
  bf16_t* packW  = (bf16_t*)alloc((size_t)C2 * CA * sizeof(bf16_t));
  float*  s_n    = (float*)alloc((size_t)NTOK * CS * 4);
  float*  Gbuf   = (float*)alloc((size_t)NTOK * CA * 4);
  float*  Sbuf   = (float*)alloc((size_t)NTOK * CA * 4);
  float*  anbuf  = (float*)alloc((size_t)NTOK * CA * 4);
  float*  sgate  = (float*)alloc((size_t)NTOK * CA * 4);
  bf16_t* qbf    = (bf16_t*)alloc((size_t)NTOK * CA * sizeof(bf16_t));
  bf16_t* kbf    = (bf16_t*)alloc((size_t)NTOK * CA * sizeof(bf16_t));
  float*  vf32   = (float*)alloc((size_t)NTOK * CA * 4);
  bf16_t* vtb    = (bf16_t*)alloc((size_t)CA * NTOK * sizeof(bf16_t));
  float*  gatef  = (float*)alloc((size_t)NTOK * CA * 4);
  float*  oheads = (float*)alloc((size_t)NTOK * CA * 4);
  float*  h1     = (float*)alloc((size_t)NTOK * C2 * 4);
  float*  h2     = (float*)alloc((size_t)NTOK * C2 * 4);
  float*  zbb    = (float*)alloc((size_t)NTOK * NTOK * NHD * 4);

  auto pack = [&](const float* src, bf16_t* dst, int R, int C) {
    dim3 g((C + 31) / 32, (R + 31) / 32);
    packT_kernel<<<g, 256, 0, stream>>>(src, dst, R, C);
  };
  auto gemm = [&](const float* A, const bf16_t* Wt, const float* bias,
                  float* oF, bf16_t* oB, const float* a1, const float* a2,
                  int K, int Nt, int emode) {
    dim3 g(Nt / TBN, NTOK / TBM);
    gemm_kernel<<<g, 256, 0, stream>>>(A, Wt, bias, oF, oB, a1, a2, NTOK, K, Nt, emode);
  };

  (void)hipMemcpyAsync(aCur, a_in, (size_t)NTOK * CA * 4, hipMemcpyDeviceToDevice, stream);

  for (int i = 0; i < 4; ++i) {
    // ---- AttentionPairBias ----
    ln_scale_kernel<<<NTOK, 128, 0, stream>>>(s_in, p_att_sg + i * CS, p_att_sb + i * CS, s_n, CS);
    pack(p_att_wg + (size_t)i * CS * CA, packW, CS, CA);
    gemm(s_n, packW, p_att_bg + i * CA, Gbuf, nullptr, nullptr, nullptr, CS, CA, 0);
    pack(p_att_ws + (size_t)i * CS * CA, packW, CS, CA);
    gemm(s_n, packW, nullptr, Sbuf, nullptr, nullptr, nullptr, CS, CA, 0);
    adaln_combine_kernel<<<NTOK, 256, 0, stream>>>(aCur, Gbuf, Sbuf, anbuf, CA);

    pack(p_ws_out + (size_t)i * CS * CA, packW, CS, CA);
    gemm(s_in, packW, p_bs_out + i * CA, sgate, nullptr, nullptr, nullptr, CS, CA, 1);

    pack(p_wq + (size_t)i * CA * CA, packW, CA, CA);
    gemm(anbuf, packW, p_bq + i * CA, nullptr, qbf, nullptr, nullptr, CA, CA, 2);
    pack(p_wk + (size_t)i * CA * CA, packW, CA, CA);
    gemm(anbuf, packW, nullptr, nullptr, kbf, nullptr, nullptr, CA, CA, 2);
    pack(p_wv + (size_t)i * CA * CA, packW, CA, CA);
    gemm(anbuf, packW, nullptr, vf32, nullptr, nullptr, nullptr, CA, CA, 0);
    pack(p_wgate + (size_t)i * CA * CA, packW, CA, CA);
    gemm(anbuf, packW, nullptr, gatef, nullptr, nullptr, nullptr, CA, CA, 0);
    pack(vf32, vtb, NTOK, CA);   // V^T bf16 [CA, NTOK]

    zbias_kernel<<<NTOK * NTOK / 8, 256, 0, stream>>>(
        z_in, p_zg + i * CZ, p_zb + i * CZ, p_wbz + (size_t)i * CZ * NHD, zbb);

    attn_kernel<<<dim3(NTOK / 16, NHD), 32, 0, stream>>>(qbf, kbf, vtb, zbb, mask, gatef, oheads);

    pack(p_wo + (size_t)i * CA * CA, packW, CA, CA);
    gemm(oheads, packW, nullptr, aCur, nullptr, sgate, aCur, CA, CA, 3);

    // ---- ConditionedTransitionBlock ----
    ln_scale_kernel<<<NTOK, 128, 0, stream>>>(s_in, p_tr_sg + i * CS, p_tr_sb + i * CS, s_n, CS);
    pack(p_tr_wg + (size_t)i * CS * CA, packW, CS, CA);
    gemm(s_n, packW, p_tr_bg + i * CA, Gbuf, nullptr, nullptr, nullptr, CS, CA, 0);
    pack(p_tr_ws + (size_t)i * CS * CA, packW, CS, CA);
    gemm(s_n, packW, nullptr, Sbuf, nullptr, nullptr, nullptr, CS, CA, 0);
    adaln_combine_kernel<<<NTOK, 256, 0, stream>>>(aCur, Gbuf, Sbuf, anbuf, CA);

    pack(p_trws_out + (size_t)i * CS * CA, packW, CS, CA);
    gemm(s_in, packW, p_trbs_out + i * CA, sgate, nullptr, nullptr, nullptr, CS, CA, 1);

    pack(p_w1 + (size_t)i * CA * C2, packW, CA, C2);
    gemm(anbuf, packW, nullptr, h1, nullptr, nullptr, nullptr, CA, C2, 4);
    pack(p_w2 + (size_t)i * CA * C2, packW, CA, C2);
    gemm(anbuf, packW, nullptr, h2, nullptr, h1, nullptr, CA, C2, 5);
    pack(p_w3 + (size_t)i * C2 * CA, packW, C2, CA);
    gemm(h2, packW, nullptr, aCur, nullptr, sgate, aCur, C2, CA, 3);
  }
}
